// TransformerGNNModel_42795054137754
// MI455X (gfx1250) — compile-verified, matrix operand order checked by reading, not run on hardware
//
#include <hip/hip_runtime.h>

#define N_NODES 2048
#define N_EDGES 65536
#define N_BATCH 32
#define DIM_IN  768
#define DIM_H   256
#define NHEAD   8
#define HEADD   32
#define ETOT    (N_EDGES + N_NODES)

typedef __attribute__((ext_vector_type(16))) __bf16 bf16x16;
typedef __attribute__((ext_vector_type(8)))  float  f32x8;

__device__ __forceinline__ __bf16 f2bf(float f) {
  unsigned u = __builtin_bit_cast(unsigned, f);
  unsigned r = u + 0x7FFFu + ((u >> 16) & 1u);          // RNE to bf16
  unsigned short h = (unsigned short)(r >> 16);
  return __builtin_bit_cast(__bf16, h);
}

__device__ __forceinline__ unsigned fenc(float f) {     // order-preserving f32->u32
  unsigned u = __builtin_bit_cast(unsigned, f);
  return (u & 0x80000000u) ? ~u : (u | 0x80000000u);
}
__device__ __forceinline__ float fdec(unsigned u) {
  u = (u & 0x80000000u) ? (u & 0x7FFFFFFFu) : ~u;
  return __builtin_bit_cast(float, u);
}

// ---------------------------------------------------------------------------
// GEMM: C[M,N] = act(A[M,K] @ W[N,K]^T + bias), bf16 WMMA, 64x64 block tile
// ---------------------------------------------------------------------------
#define LDK 40
__global__ __launch_bounds__(256) void gemm_kernel(
    const float* __restrict__ A, const float* __restrict__ W,
    const float* __restrict__ bias, float* __restrict__ C,
    int M, int N, int K, int lda, int relu)
{
  __shared__ __bf16 As[64 * LDK];
  __shared__ __bf16 Ws[64 * LDK];
  int tid  = threadIdx.x;
  int lane = tid & 31, wv = tid >> 5;
  int wm = wv & 3, wn = wv >> 2;
  int m0 = blockIdx.y * 64, n0 = blockIdx.x * 64;
  int lr = tid >> 2;           // staging row 0..63
  int lc = (tid & 3) * 8;      // staging col 0,8,16,24
  int l15 = lane & 15, kb = (lane >> 4) * 8;
  f32x8 acc0 = {}, acc1 = {};

  for (int k0 = 0; k0 < K; k0 += 32) {
    int gm = m0 + lr;
    __bf16* ad = &As[lr * LDK + lc];
    if (gm < M) {
      const float* sp = A + (size_t)gm * lda + k0 + lc;
      #pragma unroll
      for (int j = 0; j < 8; j++) ad[j] = f2bf(sp[j]);
    } else {
      #pragma unroll
      for (int j = 0; j < 8; j++) ad[j] = f2bf(0.0f);
    }
    int gn = n0 + lr;
    __bf16* wd = &Ws[lr * LDK + lc];
    if (gn < N) {
      const float* sp = W + (size_t)gn * K + k0 + lc;
      #pragma unroll
      for (int j = 0; j < 8; j++) wd[j] = f2bf(sp[j]);
    } else {
      #pragma unroll
      for (int j = 0; j < 8; j++) wd[j] = f2bf(0.0f);
    }
    __syncthreads();

    bf16x16 a, b0, b1;
    int ar  = 16 * wm + l15;
    int br0 = 32 * wn + l15;
    int br1 = br0 + 16;
    #pragma unroll
    for (int j = 0; j < 8; j++) {
      a[j]    = As[ar  * LDK + kb + j];      a[j + 8]  = As[ar  * LDK + 16 + kb + j];
      b0[j]   = Ws[br0 * LDK + kb + j];      b0[j + 8] = Ws[br0 * LDK + 16 + kb + j];
      b1[j]   = Ws[br1 * LDK + kb + j];      b1[j + 8] = Ws[br1 * LDK + 16 + kb + j];
    }
    acc0 = __builtin_amdgcn_wmma_f32_16x16x32_bf16(false, a, false, b0, (short)0, acc0, false, false);
    acc1 = __builtin_amdgcn_wmma_f32_16x16x32_bf16(false, a, false, b1, (short)0, acc1, false, false);
    __syncthreads();
  }

  int mb = m0 + 16 * wm + 8 * (lane >> 4);
  int nc0 = n0 + 32 * wn + l15;
  int nc1 = nc0 + 16;
  #pragma unroll
  for (int r = 0; r < 8; r++) {
    int m = mb + r;
    if (m < M) {
      if (nc0 < N) {
        float v = acc0[r] + (bias ? bias[nc0] : 0.0f);
        if (relu) v = fmaxf(v, 0.0f);
        C[(size_t)m * N + nc0] = v;
      }
      if (nc1 < N) {
        float v = acc1[r] + (bias ? bias[nc1] : 0.0f);
        if (relu) v = fmaxf(v, 0.0f);
        C[(size_t)m * N + nc1] = v;
      }
    }
  }
}

// ---------------------------------------------------------------------------
// Flash attention: one wave per (head, 16-query tile). d=32, S keys streamed
// in 32-key chunks. qkv layout: [tok][3*256], q at 0, k at 256, v at 512.
// ---------------------------------------------------------------------------
__global__ __launch_bounds__(32) void flash_kernel(
    const float* __restrict__ qkv, float* __restrict__ out, int S)
{
  __shared__ __bf16 Pl[16 * LDK];
  int lane = threadIdx.x;
  int h = blockIdx.x;
  int q0 = blockIdx.y * 16;
  int half = lane >> 4, l15 = lane & 15, kb = half * 8;
  const float scale = 0.17677669529663687f;   // 1/sqrt(32)

  bf16x16 qf;
  {
    const float* qp = qkv + (size_t)(q0 + l15) * 768 + h * HEADD;
    #pragma unroll
    for (int j = 0; j < 8; j++) {
      qf[j]     = f2bf(qp[kb + j] * scale);
      qf[j + 8] = f2bf(qp[16 + kb + j] * scale);
    }
  }
  f32x8 acc0 = {}, acc1 = {};
  float mrow[8], lrow[8];
  #pragma unroll
  for (int r = 0; r < 8; r++) { mrow[r] = -1e30f; lrow[r] = 0.0f; }

  for (int j0 = 0; j0 < S; j0 += 32) {
    bf16x16 k0f, k1f;
    const float* kp0 = qkv + (size_t)(j0 + l15) * 768 + 256 + h * HEADD;
    const float* kp1 = kp0 + (size_t)16 * 768;
    #pragma unroll
    for (int j = 0; j < 8; j++) {
      k0f[j] = f2bf(kp0[kb + j]); k0f[j + 8] = f2bf(kp0[16 + kb + j]);
      k1f[j] = f2bf(kp1[kb + j]); k1f[j + 8] = f2bf(kp1[16 + kb + j]);
    }
    f32x8 s0 = {}, s1 = {};
    s0 = __builtin_amdgcn_wmma_f32_16x16x32_bf16(false, qf, false, k0f, (short)0, s0, false, false);
    s1 = __builtin_amdgcn_wmma_f32_16x16x32_bf16(false, qf, false, k1f, (short)0, s1, false, false);

    #pragma unroll
    for (int r = 0; r < 8; r++) {
      float mx = fmaxf(s0[r], s1[r]);
      #pragma unroll
      for (int off = 1; off < 16; off <<= 1) mx = fmaxf(mx, __shfl_xor(mx, off, 32));
      float nm = fmaxf(mrow[r], mx);
      float fac = __expf(mrow[r] - nm);
      float p0 = __expf(s0[r] - nm), p1 = __expf(s1[r] - nm);
      s0[r] = p0; s1[r] = p1;
      float rs = p0 + p1;
      #pragma unroll
      for (int off = 1; off < 16; off <<= 1) rs += __shfl_xor(rs, off, 32);
      lrow[r] = lrow[r] * fac + rs;
      mrow[r] = nm;
      acc0[r] *= fac; acc1[r] *= fac;
    }
    // P (C layout) -> LDS -> A fragment
    #pragma unroll
    for (int r = 0; r < 8; r++) {
      Pl[(r + 8 * half) * LDK + l15]      = f2bf(s0[r]);
      Pl[(r + 8 * half) * LDK + 16 + l15] = f2bf(s1[r]);
    }
    __syncthreads();
    bf16x16 pf;
    #pragma unroll
    for (int j = 0; j < 8; j++) {
      pf[j]     = Pl[l15 * LDK + kb + j];
      pf[j + 8] = Pl[l15 * LDK + 16 + kb + j];
    }
    bf16x16 v0f, v1f;
    const float* vp = qkv + (size_t)j0 * 768 + 512 + h * HEADD;
    #pragma unroll
    for (int j = 0; j < 8; j++) {
      int key1 = kb + j, key2 = 16 + kb + j;
      v0f[j]     = f2bf(vp[(size_t)key1 * 768 + l15]);
      v0f[j + 8] = f2bf(vp[(size_t)key2 * 768 + l15]);
      v1f[j]     = f2bf(vp[(size_t)key1 * 768 + 16 + l15]);
      v1f[j + 8] = f2bf(vp[(size_t)key2 * 768 + 16 + l15]);
    }
    acc0 = __builtin_amdgcn_wmma_f32_16x16x32_bf16(false, pf, false, v0f, (short)0, acc0, false, false);
    acc1 = __builtin_amdgcn_wmma_f32_16x16x32_bf16(false, pf, false, v1f, (short)0, acc1, false, false);
    __syncthreads();
  }
  #pragma unroll
  for (int r = 0; r < 8; r++) {
    int tok = q0 + r + 8 * half;
    float inv = 1.0f / lrow[r];
    out[(size_t)tok * DIM_H + h * HEADD + l15]      = acc0[r] * inv;
    out[(size_t)tok * DIM_H + h * HEADD + 16 + l15] = acc1[r] * inv;
  }
}

// ---------------------------------------------------------------------------
// LayerNorm over 256 cols, one wave per row; out = LN(in (+res)) * g + b
// ---------------------------------------------------------------------------
__global__ __launch_bounds__(256) void ln_kernel(
    const float* __restrict__ in, const float* __restrict__ res,
    const float* __restrict__ g, const float* __restrict__ b,
    float* __restrict__ out, int M)
{
  int wid = (blockIdx.x * blockDim.x + threadIdx.x) >> 5;
  int lane = threadIdx.x & 31;
  if (wid >= M) return;
  const float* ip = in + (size_t)wid * DIM_H;
  const float* rp = res ? res + (size_t)wid * DIM_H : nullptr;
  float v[8], s = 0.0f;
  #pragma unroll
  for (int j = 0; j < 8; j++) {
    float t = ip[lane + 32 * j];
    if (rp) t += rp[lane + 32 * j];
    v[j] = t; s += t;
  }
  #pragma unroll
  for (int off = 1; off < 32; off <<= 1) s += __shfl_xor(s, off, 32);
  float mean = s * (1.0f / 256.0f);
  float vs = 0.0f;
  #pragma unroll
  for (int j = 0; j < 8; j++) { float d = v[j] - mean; vs += d * d; }
  #pragma unroll
  for (int off = 1; off < 32; off <<= 1) vs += __shfl_xor(vs, off, 32);
  float inv = rsqrtf(vs * (1.0f / 256.0f) + 1e-5f);
  #pragma unroll
  for (int j = 0; j < 8; j++) {
    int c = lane + 32 * j;
    out[(size_t)wid * DIM_H + c] = (v[j] - mean) * inv * g[c] + b[c];
  }
}

// ------------------------------ GAT kernels --------------------------------
__global__ void gat_coef_kernel(const float* __restrict__ h,
    const float* __restrict__ att_src, const float* __restrict__ att_dst,
    float* __restrict__ asrc, float* __restrict__ adst)
{
  int i = blockIdx.x * blockDim.x + threadIdx.x;
  if (i >= N_NODES * NHEAD) return;
  int node = i >> 3, hd = i & 7;
  const float* hp = h + (size_t)node * DIM_H + hd * HEADD;
  const float* sv = att_src + hd * HEADD;
  const float* dv = att_dst + hd * HEADD;
  float as = 0.0f, ad = 0.0f;
  for (int d = 0; d < HEADD; d++) { float t = hp[d]; as += t * sv[d]; ad += t * dv[d]; }
  asrc[i] = as; adst[i] = ad;
}

__global__ void gat_edge1_kernel(const int* __restrict__ ei,
    const float* __restrict__ asrc, const float* __restrict__ adst,
    float* __restrict__ ebuf, unsigned* __restrict__ emax)
{
  int i = blockIdx.x * blockDim.x + threadIdx.x;
  if (i >= ETOT * NHEAD) return;
  int e = i >> 3, hd = i & 7;
  int s = (e < N_EDGES) ? ei[e] : (e - N_EDGES);
  int d = (e < N_EDGES) ? ei[N_EDGES + e] : (e - N_EDGES);
  float v = asrc[s * NHEAD + hd] + adst[d * NHEAD + hd];
  v = (v > 0.0f) ? v : 0.2f * v;                // leaky_relu 0.2
  ebuf[i] = v;
  atomicMax(&emax[d * NHEAD + hd], fenc(v));
}

__global__ void gat_edge2_kernel(const int* __restrict__ ei,
    const unsigned* __restrict__ emax, float* __restrict__ ebuf,
    float* __restrict__ denom)
{
  int i = blockIdx.x * blockDim.x + threadIdx.x;
  if (i >= ETOT * NHEAD) return;
  int e = i >> 3, hd = i & 7;
  int d = (e < N_EDGES) ? ei[N_EDGES + e] : (e - N_EDGES);
  float ex = __expf(ebuf[i] - fdec(emax[d * NHEAD + hd]));
  ebuf[i] = ex;
  atomicAdd(&denom[d * NHEAD + hd], ex);
}

__global__ void gat_scatter_kernel(const int* __restrict__ ei,
    const float* __restrict__ ebuf, const float* __restrict__ denom,
    const float* __restrict__ h, float* __restrict__ out)
{
  int i = blockIdx.x * blockDim.x + threadIdx.x;
  if (i >= ETOT * NHEAD) return;
  int e = i >> 3, hd = i & 7;
  int s = (e < N_EDGES) ? ei[e] : (e - N_EDGES);
  int d = (e < N_EDGES) ? ei[N_EDGES + e] : (e - N_EDGES);
  float alpha = ebuf[i] / denom[d * NHEAD + hd];
  const float* hp = h + (size_t)s * DIM_H + hd * HEADD;
  float* op = out + (size_t)d * DIM_H + hd * HEADD;
  for (int k = 0; k < HEADD; k++) atomicAdd(&op[k], alpha * hp[k]);
}

__global__ void bias_relu_kernel(float* __restrict__ x,
    const float* __restrict__ bias, int M, int Nc)
{
  int i = blockIdx.x * blockDim.x + threadIdx.x;
  if (i >= M * Nc) return;
  x[i] = fmaxf(x[i] + bias[i % Nc], 0.0f);
}

// ------------------------------ pooling ------------------------------------
__global__ void pool_sum_kernel(const float* __restrict__ xg,
    const int* __restrict__ batch, float* __restrict__ pooled,
    float* __restrict__ cnt)
{
  int i = blockIdx.x * blockDim.x + threadIdx.x;
  if (i >= N_NODES * DIM_H) return;
  int node = i >> 8, c = i & 255;
  int bb = batch[node];
  atomicAdd(&pooled[bb * DIM_H + c], xg[i]);
  if (c == 0) atomicAdd(&cnt[bb], 1.0f);
}
__global__ void pool_div_kernel(const float* __restrict__ pooled,
    const float* __restrict__ cnt, float* __restrict__ xt)
{
  int i = blockIdx.x * blockDim.x + threadIdx.x;
  if (i >= N_BATCH * DIM_H) return;
  xt[i] = pooled[i] / fmaxf(cnt[i >> 8], 1.0f);
}
__global__ void wpool_kernel(const float* __restrict__ xg,
    const float* __restrict__ w, const int* __restrict__ batch,
    float* __restrict__ xw, float* __restrict__ wsum)
{
  int i = blockIdx.x * blockDim.x + threadIdx.x;
  if (i >= N_NODES * DIM_H) return;
  int node = i >> 8, c = i & 255;
  int bb = batch[node];
  atomicAdd(&xw[bb * DIM_H + c], xg[i] * w[node]);
  if (c == 0) atomicAdd(&wsum[bb], w[node]);
}
__global__ void xw_norm_kernel(float* __restrict__ xw,
    const float* __restrict__ wsum)
{
  int i = blockIdx.x * blockDim.x + threadIdx.x;
  if (i >= N_BATCH * DIM_H) return;
  float s = wsum[i >> 8];
  xw[i] = (s > 0.0f) ? xw[i] / s : 0.0f;
}

// ------------------------- small-N gemm / concat ---------------------------
__global__ void smallgemm_kernel(const float* __restrict__ A,
    const float* __restrict__ W, const float* __restrict__ bias,
    float* __restrict__ C, int M, int Nc, int K, int act)
{
  int i = blockIdx.x * blockDim.x + threadIdx.x;
  if (i >= M * Nc) return;
  int m = i / Nc, nn = i % Nc;
  const float* a = A + (size_t)m * K;
  const float* wv = W + (size_t)nn * K;
  float s = bias ? bias[nn] : 0.0f;
  for (int k = 0; k < K; k++) s += a[k] * wv[k];
  if (act == 1) s = fmaxf(s, 0.0f);
  else if (act == 2) s = 1.0f / (1.0f + __expf(-s));
  C[i] = s;
}
__global__ void concat_kernel(const float* __restrict__ a,
    const float* __restrict__ b, float* __restrict__ out)
{
  int i = blockIdx.x * blockDim.x + threadIdx.x;
  if (i >= N_BATCH * 2 * DIM_H) return;
  int m = i >> 9, c = i & 511;
  out[i] = (c < DIM_H) ? a[m * DIM_H + c] : b[m * DIM_H + (c - DIM_H)];
}

// ---------------------------------------------------------------------------
extern "C" void kernel_launch(void* const* d_in, const int* in_sizes, int n_in,
                              void* d_out, int out_size, void* d_ws, size_t ws_size,
                              hipStream_t stream)
{
  (void)in_sizes; (void)n_in; (void)out_size; (void)ws_size;
  const float* x_in  = (const float*)d_in[0];
  const int*   ei    = (const int*)d_in[1];
  const int*   batch = (const int*)d_in[2];

  int idx = 3;
  auto nx = [&]() { return (const float*)d_in[idx++]; };
  // params flattened in jax pytree order (sorted dict keys)
  const float *cls1W = nx(), *cls1b = nx();
  const float *cls2W = nx(), *cls2b = nx();
  const float *cls3W = nx(), *cls3b = nx();
  const float *gatt1W = nx(), *gatt1b = nx();
  const float *gatt2W = nx(), *gatt2b = nx();
  const float *gmWin = nx(), *gmWo = nx(), *gmbin = nx(), *gmbo = nx();
  const float *inpW = nx(), *inpb = nx(), *inplnb = nx(), *inplng = nx();
  struct Layer {
    const float *ff1W,*ff1b,*ff2W,*ff2b;
    const float *gatW,*attdst,*attsrc,*gatbias;
    const float *mWin,*mWo,*mbin,*mbo;
    const float *n1b,*n1g,*n2b,*n2g,*n3b,*n3g;
  } L[4];
  for (int l = 0; l < 4; l++) {
    L[l].ff1W = nx(); L[l].ff1b = nx();
    L[l].ff2W = nx(); L[l].ff2b = nx();
    L[l].gatW = nx(); L[l].attdst = nx(); L[l].attsrc = nx(); L[l].gatbias = nx();
    L[l].mWin = nx(); L[l].mWo = nx(); L[l].mbin = nx(); L[l].mbo = nx();
    L[l].n1b = nx(); L[l].n1g = nx();
    L[l].n2b = nx(); L[l].n2g = nx();
    L[l].n3b = nx(); L[l].n3g = nx();
  }
  struct Temp {
    const float *ff1W,*ff1b,*ff2W,*ff2b;
    const float *mWin,*mWo,*mbin,*mbo;
    const float *n1b,*n1g,*n2b,*n2g;
  } T[2];
  for (int t = 0; t < 2; t++) {
    T[t].ff1W = nx(); T[t].ff1b = nx();
    T[t].ff2W = nx(); T[t].ff2b = nx();
    T[t].mWin = nx(); T[t].mWo = nx(); T[t].mbin = nx(); T[t].mbo = nx();
    T[t].n1b = nx(); T[t].n1g = nx();
    T[t].n2b = nx(); T[t].n2g = nx();
  }
  const float *tp1W = nx(), *tp1b = nx();
  const float *tp2W = nx(), *tp2b = nx();

  // workspace carve
  float* wp = (float*)d_ws;
  float* X    = wp; wp += (size_t)N_NODES * DIM_H;
  float* XG   = wp; wp += (size_t)N_NODES * DIM_H;
  float* BIG  = wp; wp += (size_t)N_NODES * 4 * DIM_H;
  float* TMP  = wp; wp += (size_t)N_NODES * DIM_H;
  float* QKV  = wp; wp += (size_t)N_NODES * 3 * DIM_H;
  float* ASRC = wp; wp += (size_t)N_NODES * NHEAD;
  float* ADST = wp; wp += (size_t)N_NODES * NHEAD;
  float* EBUF = wp; wp += (size_t)ETOT * NHEAD;
  unsigned* EMAX = (unsigned*)wp; wp += (size_t)N_NODES * NHEAD;
  float* DEN  = wp; wp += (size_t)N_NODES * NHEAD;
  float* POOL = wp; wp += N_BATCH * DIM_H;
  float* CNT  = wp; wp += N_BATCH;
  float* XT   = wp; wp += N_BATCH * DIM_H;
  float* XTA  = wp; wp += N_BATCH * DIM_H;
  float* QKVT = wp; wp += N_BATCH * 3 * DIM_H;
  float* FFB  = wp; wp += N_BATCH * 2 * DIM_H;
  float* WG   = wp; wp += N_NODES;
  float* WSUM = wp; wp += N_BATCH;
  float* XW   = wp; wp += N_BATCH * DIM_H;
  float* XC   = wp; wp += N_BATCH * 2 * DIM_H;
  float* C1   = wp; wp += N_BATCH * DIM_H;
  float* C2   = wp; wp += N_BATCH * DIM_H / 2;
  float* T1O  = wp; wp += N_BATCH * DIM_H / 2;

  auto gemm = [&](const float* A, const float* W, const float* bias, float* C,
                  int M, int Ncol, int K, int lda, int relu) {
    dim3 g((Ncol + 63) / 64, (M + 63) / 64);
    gemm_kernel<<<g, 256, 0, stream>>>(A, W, bias, C, M, Ncol, K, lda, relu);
  };
  auto ln = [&](const float* in, const float* res, const float* g,
                const float* b, float* out, int M) {
    ln_kernel<<<(M * 32 + 255) / 256, 256, 0, stream>>>(in, res, g, b, out, M);
  };
  const int EG = (ETOT * NHEAD + 255) / 256;
  const int NG = (N_NODES * DIM_H + 255) / 256;

  // input projection: x = LN(relu(x_in @ inpW^T + inpb))
  gemm(x_in, inpW, inpb, TMP, N_NODES, DIM_H, DIM_IN, DIM_IN, 1);
  ln(TMP, nullptr, inplng, inplnb, X, N_NODES);

  for (int l = 0; l < 4; l++) {
    // GAT
    gemm(X, L[l].gatW, nullptr, BIG, N_NODES, DIM_H, DIM_H, DIM_H, 0);
    gat_coef_kernel<<<(N_NODES * NHEAD + 255) / 256, 256, 0, stream>>>(
        BIG, L[l].attsrc, L[l].attdst, ASRC, ADST);
    hipMemsetAsync(EMAX, 0, (size_t)N_NODES * NHEAD * 4, stream);
    hipMemsetAsync(DEN,  0, (size_t)N_NODES * NHEAD * 4, stream);
    hipMemsetAsync(TMP,  0, (size_t)N_NODES * DIM_H * 4, stream);
    gat_edge1_kernel<<<EG, 256, 0, stream>>>(ei, ASRC, ADST, EBUF, EMAX);
    gat_edge2_kernel<<<EG, 256, 0, stream>>>(ei, EMAX, EBUF, DEN);
    gat_scatter_kernel<<<EG, 256, 0, stream>>>(ei, EBUF, DEN, BIG, TMP);
    bias_relu_kernel<<<NG, 256, 0, stream>>>(TMP, L[l].gatbias, N_NODES, DIM_H);
    ln(TMP, X, L[l].n1g, L[l].n1b, X, N_NODES);
    // MHA
    gemm(X, L[l].mWin, L[l].mbin, QKV, N_NODES, 3 * DIM_H, DIM_H, DIM_H, 0);
    flash_kernel<<<dim3(NHEAD, N_NODES / 16), 32, 0, stream>>>(QKV, TMP, N_NODES);
    gemm(TMP, L[l].mWo, L[l].mbo, BIG, N_NODES, DIM_H, DIM_H, DIM_H, 0);
    ln(BIG, X, L[l].n2g, L[l].n2b, X, N_NODES);
    // FF
    gemm(X, L[l].ff1W, L[l].ff1b, BIG, N_NODES, 4 * DIM_H, DIM_H, DIM_H, 1);
    gemm(BIG, L[l].ff2W, L[l].ff2b, TMP, N_NODES, DIM_H, 4 * DIM_H, 4 * DIM_H, 0);
    ln(TMP, X, L[l].n3g, L[l].n3b, X, N_NODES);
  }

  // global MHA: xg = mha(x) (no residual/LN)
  gemm(X, gmWin, gmbin, QKV, N_NODES, 3 * DIM_H, DIM_H, DIM_H, 0);
  flash_kernel<<<dim3(NHEAD, N_NODES / 16), 32, 0, stream>>>(QKV, TMP, N_NODES);
  gemm(TMP, gmWo, gmbo, XG, N_NODES, DIM_H, DIM_H, DIM_H, 0);

  // segment mean pool -> xt
  hipMemsetAsync(POOL, 0, (size_t)N_BATCH * DIM_H * 4, stream);
  hipMemsetAsync(CNT,  0, (size_t)N_BATCH * 4, stream);
  pool_sum_kernel<<<NG, 256, 0, stream>>>(XG, batch, POOL, CNT);
  pool_div_kernel<<<(N_BATCH * DIM_H + 255) / 256, 256, 0, stream>>>(POOL, CNT, XT);

  // temporal transformer layers (S=1 => attention output == V)
  for (int t = 0; t < 2; t++) {
    gemm(XT, T[t].mWin, T[t].mbin, QKVT, N_BATCH, 3 * DIM_H, DIM_H, DIM_H, 0);
    gemm(QKVT + 2 * DIM_H, T[t].mWo, T[t].mbo, XTA, N_BATCH, DIM_H, DIM_H, 3 * DIM_H, 0);
    ln(XTA, XT, T[t].n1g, T[t].n1b, XT, N_BATCH);
    gemm(XT, T[t].ff1W, T[t].ff1b, FFB, N_BATCH, 2 * DIM_H, DIM_H, DIM_H, 1);
    gemm(FFB, T[t].ff2W, T[t].ff2b, XTA, N_BATCH, DIM_H, 2 * DIM_H, 2 * DIM_H, 0);
    ln(XTA, XT, T[t].n2g, T[t].n2b, XT, N_BATCH);
  }

  // gated pooling weights
  gemm(XG, gatt1W, gatt1b, BIG, N_NODES, DIM_H / 2, DIM_H, DIM_H, 1);
  smallgemm_kernel<<<(N_NODES + 255) / 256, 256, 0, stream>>>(
      BIG, gatt2W, gatt2b, WG, N_NODES, 1, DIM_H / 2, 2);
  hipMemsetAsync(XW,   0, (size_t)N_BATCH * DIM_H * 4, stream);
  hipMemsetAsync(WSUM, 0, (size_t)N_BATCH * 4, stream);
  wpool_kernel<<<NG, 256, 0, stream>>>(XG, WG, batch, XW, WSUM);
  xw_norm_kernel<<<(N_BATCH * DIM_H + 255) / 256, 256, 0, stream>>>(XW, WSUM);

  // classifier head
  concat_kernel<<<(N_BATCH * 2 * DIM_H + 255) / 256, 256, 0, stream>>>(XT, XW, XC);
  gemm(XC, cls1W, cls1b, C1, N_BATCH, DIM_H, 2 * DIM_H, 2 * DIM_H, 1);
  gemm(C1, cls2W, cls2b, C2, N_BATCH, DIM_H / 2, DIM_H, DIM_H, 1);
  smallgemm_kernel<<<1, 256, 0, stream>>>(
      C2, cls3W, cls3b, (float*)d_out, N_BATCH, 4, DIM_H / 2, 0);

  // temporal importance head
  gemm(XT, tp1W, tp1b, T1O, N_BATCH, DIM_H / 2, DIM_H, DIM_H, 1);
  smallgemm_kernel<<<1, 256, 0, stream>>>(
      T1O, tp2W, tp2b, (float*)d_out + N_BATCH * 4, N_BATCH, 1, DIM_H / 2, 2);
}
